// EdgeMask_34488587387482
// MI455X (gfx1250) — matrix operand order repaired and checked by reading, not compiled, not run on hardware
//
#include <hip/hip_runtime.h>

typedef __attribute__((ext_vector_type(16))) _Float16 v16h;
typedef __attribute__((ext_vector_type(8)))  _Float16 v8h;
typedef __attribute__((ext_vector_type(8)))  float    v8f;

#define NTOK    128   // N
#define DMODEL  128   // d
#define HID     32    // hidden
#define HID2    64    // 2*hidden (li | lj columns)
#define ASTRIDE 136   // halves per row of staged h tile (272B: 16B-aligned, 17 super-banks)
#define BSTRIDE 136   // halves per column of staged W (column-major)
#define LSTRIDE 68    // floats per row of L (272B: 16B-aligned, conflict-free f4 reads)

__global__ __launch_bounds__(256) void edgemask_kernel(
    const float* __restrict__ h,
    const float* __restrict__ I_full,
    const float* __restrict__ W1,
    const float* __restrict__ b1,
    const float* __restrict__ W2,
    const float* __restrict__ b2,
    float* __restrict__ outI,
    float* __restrict__ outM)
{
  __shared__ __align__(16) _Float16 sA[NTOK * ASTRIDE];   // h tile, f16, row-major
  __shared__ __align__(16) _Float16 sB[HID2 * BSTRIDE];   // W tile, f16, COLUMN-major: sB[n*136+k]
  __shared__ __align__(16) float    sL[NTOK * LSTRIDE];   // L = h@W (+b1 on li half), f32
  __shared__ float sb1[HID];
  __shared__ float sW2[HID];

  const int t    = threadIdx.x;
  const int lane = t & 31;
  const int wv   = t >> 5;                   // 8 waves
  const int bt   = blockIdx.x;               // one (b,t) per block

  const float* __restrict__ hb = h + (size_t)bt * NTOK * DMODEL;

  // ---------- Phase 0: stage h (f32->f16) and fused W1 into LDS ----------
  #pragma unroll
  for (int it = 0; it < 16; ++it) {
    int idx4 = t + 256 * it;                 // 4096 float4's
    int flat = idx4 * 4;
    int row  = flat >> 7;
    int col  = flat & 127;
    float4 v = ((const float4*)hb)[idx4];
    _Float16* dst = &sA[row * ASTRIDE + col];
    dst[0] = (_Float16)v.x; dst[1] = (_Float16)v.y;
    dst[2] = (_Float16)v.z; dst[3] = (_Float16)v.w;
  }
  #pragma unroll
  for (int it = 0; it < 32; ++it) {
    int idx = t + 256 * it;                  // 8192 = 64 cols x 128 k
    int n = idx >> 7;                        // output column 0..63
    int k = idx & 127;                       // reduction index
    float w = (n < HID) ? W1[k * HID + n] : W1[(DMODEL + k) * HID + (n - HID)];
    sB[n * BSTRIDE + k] = (_Float16)w;       // column-major: lane-contiguous fragments
  }
  if (t < HID) { sb1[t] = b1[t]; sW2[t] = W2[t]; }
  __syncthreads();

  // ---------- Phase 1: L(128x64) = h(128x128) @ W(128x64) via WMMA ----------
  const int rowBase = wv * 16;               // wave's 16-row tile
  const int mrow    = rowBase + (lane & 15);
  const int hiHalf  = (lane >= 16) ? 1 : 0;
  const int aBase   = hiHalf ? 8 : 0;        // A: lanes 16-31 hold K+8 group
  const int bBase   = hiHalf ? 8 : 0;        // B: lanes 16-31 hold K+8 group

  v8f acc[4] = { v8f{}, v8f{}, v8f{}, v8f{} };

  #pragma unroll
  for (int kk = 0; kk < DMODEL; kk += 32) {
    // A fragment: halves 0..7 = K[kk+aBase .. +8), halves 8..15 = K[kk+16+aBase .. +8)
    const v8h a0 = *(const v8h*)&sA[mrow * ASTRIDE + kk + aBase];
    const v8h a1 = *(const v8h*)&sA[mrow * ASTRIDE + kk + 16 + aBase];
    const v16h a = __builtin_shufflevector(a0, a1,
        0,1,2,3,4,5,6,7,8,9,10,11,12,13,14,15);
    #pragma unroll
    for (int nt = 0; nt < 4; ++nt) {
      // B fragment: halves 0..7 = K[kk+bBase .. +8), halves 8..15 = K[kk+16+bBase .. +8)
      const _Float16* pb = &sB[(nt * 16 + (lane & 15)) * BSTRIDE + kk + bBase];
      const v8h b0 = *(const v8h*)(pb);
      const v8h b1v = *(const v8h*)(pb + 16);
      const v16h b = __builtin_shufflevector(b0, b1v,
          0,1,2,3,4,5,6,7,8,9,10,11,12,13,14,15);
      acc[nt] = __builtin_amdgcn_wmma_f32_16x16x32_f16(
          false, a, false, b, (short)0, acc[nt], false, false);
    }
  }

  // Write back D tiles; fold b1 into the li (n<32) half
  #pragma unroll
  for (int nt = 0; nt < 4; ++nt) {
    #pragma unroll
    for (int r = 0; r < 8; ++r) {
      int i = rowBase + r + (hiHalf ? 8 : 0);
      int n = nt * 16 + (lane & 15);
      float v = acc[nt][r];
      if (n < HID) v += sb1[n];
      sL[i * LSTRIDE + n] = v;
    }
  }
  __syncthreads();

  // ---------- Phase 2: pairwise MLP + sigmoid + mask ----------
  float w2r[HID];
  #pragma unroll
  for (int k = 0; k < HID; ++k) w2r[k] = sW2[k];
  const float b2v = b2[0];
  const size_t base = (size_t)bt * NTOK * NTOK;

  #pragma unroll 1
  for (int jb = 0; jb < 4; ++jb) {
    const int j = jb * 32 + lane;            // lane-contiguous j: coalesced I/O
    float ljr[HID];
    #pragma unroll
    for (int k4 = 0; k4 < HID; k4 += 4) {
      float4 v = *(const float4*)&sL[j * LSTRIDE + HID + k4];
      ljr[k4+0] = v.x; ljr[k4+1] = v.y; ljr[k4+2] = v.z; ljr[k4+3] = v.w;
    }
    #pragma unroll 4
    for (int it = 0; it < 16; ++it) {
      const int i = wv * 16 + it;            // uniform per wave: LDS broadcast
      float acc2 = b2v;
      #pragma unroll
      for (int k4 = 0; k4 < HID; k4 += 4) {
        float4 li4 = *(const float4*)&sL[i * LSTRIDE + k4];
        float t0 = li4.x + ljr[k4+0]; t0 = t0 > 0.f ? t0 : 0.f; acc2 = fmaf(t0, w2r[k4+0], acc2);
        float t1 = li4.y + ljr[k4+1]; t1 = t1 > 0.f ? t1 : 0.f; acc2 = fmaf(t1, w2r[k4+1], acc2);
        float t2 = li4.z + ljr[k4+2]; t2 = t2 > 0.f ? t2 : 0.f; acc2 = fmaf(t2, w2r[k4+2], acc2);
        float t3 = li4.w + ljr[k4+3]; t3 = t3 > 0.f ? t3 : 0.f; acc2 = fmaf(t3, w2r[k4+3], acc2);
      }
      const float m = 1.0f / (1.0f + __expf(-acc2));
      const size_t off = base + (size_t)i * NTOK + j;
      const float Iv = I_full[off];
      outI[off] = Iv * m;
      outM[off] = m;
    }
  }
}

extern "C" void kernel_launch(void* const* d_in, const int* in_sizes, int n_in,
                              void* d_out, int out_size, void* d_ws, size_t ws_size,
                              hipStream_t stream) {
  const float* h  = (const float*)d_in[0];
  const float* I  = (const float*)d_in[1];
  const float* W1 = (const float*)d_in[2];
  const float* b1 = (const float*)d_in[3];
  const float* W2 = (const float*)d_in[4];
  const float* b2 = (const float*)d_in[5];

  const size_t NOUT = (size_t)8 * 32 * 128 * 128;   // B*T*N*N
  float* outI = (float*)d_out;
  float* outM = outI + NOUT;

  dim3 grid(256), block(256);                       // one block per (b,t)
  hipLaunchKernelGGL(edgemask_kernel, grid, block, 0, stream,
                     h, I, W1, b1, W2, b2, outI, outM);
}
